// MuonGatedDeltaNetVisionMixer_61641370632619
// MI455X (gfx1250) — matrix operand order repaired and checked by reading, not compiled, not run on hardware
//
#include <hip/hip_runtime.h>
#include <hip/hip_bf16.h>
#include <stdint.h>

typedef unsigned short u16;
typedef __attribute__((ext_vector_type(16))) __bf16 v16bf;
typedef __attribute__((ext_vector_type(8)))  float  v8f;

union FragBF { uint4 u[2]; v16bf v; };

// Problem constants
#define NB   8
#define NL   2048
#define DM   1024
#define DI   2048
#define DSSM 1024
#define NH   4
#define DK   192
#define DV   384
#define ROWS (NB * NL)       // 16384
#define NCAT 4608            // q(768) k(768) v(1536) g(1536)

// ---------------------------------------------------------------------------
// CDNA5 async global->LDS copy (ASYNCcnt-tracked) + wait.
// LDS operand is the 32-bit LDS byte offset (low 32 bits of the flat address).
// ---------------------------------------------------------------------------
__device__ __forceinline__ unsigned lds_off(const void* p) {
  return (unsigned)(uintptr_t)p;
}
__device__ __forceinline__ void async_b128(unsigned lds, const void* g) {
  asm volatile("global_load_async_to_lds_b128 %0, %1, off"
               :: "v"(lds), "v"((unsigned long long)(uintptr_t)g)
               : "memory");
}
__device__ __forceinline__ void wait_async0() {
  asm volatile("s_wait_asynccnt 0" ::: "memory");
}

// ---------------------------------------------------------------------------
// f32 -> bf16 slice cast (used for weights / activations feeding WMMA)
// ---------------------------------------------------------------------------
__global__ void cast_f32_to_bf16_slice(const float* __restrict__ src,
                                       __hip_bfloat16* __restrict__ dst,
                                       long long n, int src_cols, int dst_ld, int dst_off) {
  long long idx = (long long)blockIdx.x * blockDim.x + threadIdx.x;
  if (idx >= n) return;
  long long row = idx / src_cols;
  int col = (int)(idx - row * src_cols);
  dst[row * dst_ld + dst_off + col] = __float2bfloat16(src[idx]);
}

// ---------------------------------------------------------------------------
// bf16 WMMA GEMM: C[M,N] = A[M,K] * B[K,N], f32 accumulate.
// Block tile 128x128 (256 threads, 8 waves -> 4x2 waves of 32x64).
// Double-buffered LDS; A tile staged with global_load_async_to_lds_b128,
// B tile register-prefetched and transposed into LDS during WMMA compute.
// LDS rows padded to stride 40 (bank-conflict-free, 16B aligned).
// Requires M%128==0, N%128==0, K%32==0 (true for every GEMM here).
// ---------------------------------------------------------------------------
template <bool OUT_BF16>
__global__ __launch_bounds__(256) void gemm_bf16_wmma(
    const u16* __restrict__ A, const u16* __restrict__ Bm,
    int M, int N, int K, void* __restrict__ Cout, int ldc, int coff) {
  __shared__ __align__(16) u16 As[2][128 * 40];
  __shared__ __align__(16) u16 Bs[2][128 * 40];  // transposed: Bs[n][k]

  const int tid  = threadIdx.x;
  const int wave = tid >> 5;
  const int lane = tid & 31;
  const int mw = wave & 3;      // 4 waves along M (32 rows each)
  const int nw = wave >> 2;     // 2 waves along N (64 cols each)
  const int lm = lane & 15;
  const int lk = lane >> 4;     // K-half select per ISA 16-bit layout
  const int mBase = blockIdx.x * 128;
  const int nBase = blockIdx.y * 128;

  v8f zero;
#pragma unroll
  for (int q = 0; q < 8; ++q) zero[q] = 0.f;
  v8f acc[2][4];
#pragma unroll
  for (int i = 0; i < 2; ++i)
#pragma unroll
    for (int j = 0; j < 4; ++j) acc[i][j] = zero;

  const int ar = tid >> 1;   // A stage: row 0..127
  const int ah = tid & 1;    // A stage: 16-elem half

  // ---- helpers -------------------------------------------------------------
  auto stageA_async = [&](int buf, int kb) {
    unsigned dst = lds_off(&As[buf][ar * 40 + ah * 16]);
    const u16* src = A + (size_t)(mBase + ar) * K + kb + ah * 16;
    async_b128(dst, src);                // 16B
    async_b128(dst + 16, src + 8);       // 16B
  };
  unsigned pairs[8];
  auto loadB = [&](int kb) {
#pragma unroll
    for (int i = 0; i < 8; ++i) {
      int e = tid + (i << 8);            // 0..2047 (uint pairs)
      int k = e >> 6;                    // 0..31
      int n2 = (e & 63) << 1;            // even n
      pairs[i] = *(const unsigned int*)(Bm + (size_t)(kb + k) * N + nBase + n2);
    }
  };
  auto storeB = [&](int buf) {
#pragma unroll
    for (int i = 0; i < 8; ++i) {
      int e = tid + (i << 8);
      int k = e >> 6;
      int n2 = (e & 63) << 1;
      Bs[buf][(n2)*40 + k]     = (u16)(pairs[i] & 0xffffu);
      Bs[buf][(n2 + 1)*40 + k] = (u16)(pairs[i] >> 16);
    }
  };

  // ---- prologue: stage tile 0 ---------------------------------------------
  loadB(0);
  stageA_async(0, 0);
  storeB(0);
  wait_async0();
  __syncthreads();

  const int nsteps = K >> 5;
  for (int s = 0; s < nsteps; ++s) {
    const int cur = s & 1;
    const bool hasNext = (s + 1 < nsteps);
    if (hasNext) loadB((s + 1) << 5);    // global B prefetch overlaps WMMA

    FragBF a[2], b[4];
#pragma unroll
    for (int mi = 0; mi < 2; ++mi) {
      int m = mw * 32 + mi * 16 + lm;
      a[mi].u[0] = *(const uint4*)(&As[cur][m * 40 + lk * 8]);       // K = lk*8..+7
      a[mi].u[1] = *(const uint4*)(&As[cur][m * 40 + 16 + lk * 8]);  // K = 16+lk*8..+7
    }
#pragma unroll
    for (int ni = 0; ni < 4; ++ni) {
      int n = nw * 64 + ni * 16 + lm;
      b[ni].u[0] = *(const uint4*)(&Bs[cur][n * 40 + lk * 8]);
      b[ni].u[1] = *(const uint4*)(&Bs[cur][n * 40 + 16 + lk * 8]);
    }
#pragma unroll
    for (int mi = 0; mi < 2; ++mi)
#pragma unroll
      for (int ni = 0; ni < 4; ++ni)
        acc[mi][ni] = __builtin_amdgcn_wmma_f32_16x16x32_bf16(
            false, a[mi].v, false, b[ni].v, (short)0, acc[mi][ni], false, false);

    if (hasNext) {
      stageA_async(cur ^ 1, (s + 1) << 5);
      storeB(cur ^ 1);
      wait_async0();
    }
    __syncthreads();
  }

  // Epilogue: C/D layout — VGPR r: lanes 0-15 -> M=r, lanes 16-31 -> M=8+r; N=lane&15
#pragma unroll
  for (int mi = 0; mi < 2; ++mi)
#pragma unroll
    for (int ni = 0; ni < 4; ++ni)
#pragma unroll
      for (int rr = 0; rr < 8; ++rr) {
        int m = mBase + mw * 32 + mi * 16 + lk * 8 + rr;
        int n = nBase + nw * 64 + ni * 16 + lm;
        float val = acc[mi][ni][rr];
        if (OUT_BF16)
          ((__hip_bfloat16*)Cout)[(size_t)m * ldc + coff + n] = __float2bfloat16(val);
        else
          ((float*)Cout)[(size_t)m * ldc + coff + n] = val;
      }
}

// ---------------------------------------------------------------------------
// Depthwise conv (K=4, pad left 1 / right 2) + SiLU.  xz is [ROWS, DI] f32.
// ---------------------------------------------------------------------------
__global__ void conv1d_silu(const float* __restrict__ xz, int chanOff,
                            const float* __restrict__ w, const float* __restrict__ bias,
                            float* __restrict__ outF, __hip_bfloat16* __restrict__ outB,
                            int bld, int boff) {
  long long idx = (long long)blockIdx.x * blockDim.x + threadIdx.x;
  if (idx >= (long long)ROWS * 1024) return;
  int c = (int)(idx & 1023);
  int l = (int)((idx >> 10) & 2047);
  int b = (int)(idx >> 21);
  const float* col = xz + (size_t)b * NL * DI + chanOff + c;
  const float* wc = w + c * 4;
  float acc = bias[c];
#pragma unroll
  for (int j = 0; j < 4; ++j) {
    int ls = l - 1 + j;
    if (ls >= 0 && ls < NL) acc = fmaf(wc[j], col[(size_t)ls * DI], acc);
  }
  float s = acc / (1.f + __expf(-acc));  // SiLU
  size_t rowi = (size_t)b * NL + l;
  if (outF) outF[rowi * 1024 + c] = s;
  outB[rowi * bld + boff + c] = __float2bfloat16(s);
}

// ---------------------------------------------------------------------------
// gk = -exp(A_log)*softplus(xs@W_gk + dt_bias);  beta = sigmoid(xs@W_b + b_b)
// One wave per (row, h, gk|beta) output, f32 dot of length 1024.
// ---------------------------------------------------------------------------
__global__ __launch_bounds__(256) void gk_beta_kernel(
    const float* __restrict__ xs, const float* __restrict__ Wgk, const float* __restrict__ Wb,
    const float* __restrict__ dtb, const float* __restrict__ b_b, const float* __restrict__ A_log,
    float* __restrict__ gkOut, float* __restrict__ betaOut) {
  int wave = threadIdx.x >> 5, lane = threadIdx.x & 31;
  long long unit = (long long)blockIdx.x * 8 + wave;  // ROWS*8 units
  int row = (int)(unit >> 3);
  int rem = (int)(unit & 7);
  int h = rem & 3;
  int isBeta = rem >> 2;
  const float* W = isBeta ? Wb : Wgk;
  const float* xr = xs + (size_t)row * 1024;
  float acc = 0.f;
  for (int d = lane; d < 1024; d += 32) acc = fmaf(xr[d], W[d * 4 + h], acc);
#pragma unroll
  for (int off = 16; off > 0; off >>= 1) acc += __shfl_xor(acc, off, 32);
  if (lane == 0) {
    int b = row >> 11, l = row & 2047;
    size_t o = ((size_t)b * NH + h) * NL + l;
    if (isBeta) {
      betaOut[o] = 1.f / (1.f + __expf(-(acc + b_b[h])));
    } else {
      float xv = acc + dtb[h];
      float sp = (xv > 20.f) ? xv : log1pf(__expf(xv));
      gkOut[o] = -__expf(A_log[h]) * sp;
    }
  }
}

// ---------------------------------------------------------------------------
// In-place L2 norm of q and k head-slices inside qkvg [ROWS, 4608] f32.
// ---------------------------------------------------------------------------
__global__ __launch_bounds__(256) void l2norm_qk(float* __restrict__ qkvg) {
  int wave = threadIdx.x >> 5, lane = threadIdx.x & 31;
  long long unit = (long long)blockIdx.x * 8 + wave;  // ROWS*8
  int row = (int)(unit >> 3);
  int rem = (int)(unit & 7);
  int which = rem >> 2;  // 0=q, 1=k
  int h = rem & 3;
  float* p = qkvg + (size_t)row * NCAT + which * 768 + h * DK;
  float v[6];
  float ss = 0.f;
#pragma unroll
  for (int i = 0; i < 6; ++i) { v[i] = p[lane + i * 32]; ss = fmaf(v[i], v[i], ss); }
#pragma unroll
  for (int off = 16; off > 0; off >>= 1) ss += __shfl_xor(ss, off, 32);
  float sc = rsqrtf(ss + 1e-6f);
#pragma unroll
  for (int i = 0; i < 6; ++i) p[lane + i * 32] = v[i] * sc;
}

// ---------------------------------------------------------------------------
// Gated delta rule recurrent scan (exact f32, matches reference semantics).
// Grid: (B*H)*SCAN_G blocks; block = 64 threads, each lane owns one DV column
// and keeps its 192-element state column in registers.  q/k/v staged through
// double-buffered LDS chunks entirely via async b128 copies (overlapped with
// the 16-step compute of the previous chunk); k/q consumed as broadcast
// float4 LDS loads.
// ---------------------------------------------------------------------------
#define SCAN_T  16
#define SCAN_G  6   // 6 column groups of 64 -> DV=384
#define SCAN_CW 64

__global__ __launch_bounds__(64) void gated_delta_scan(
    const float* __restrict__ qkvg, const float* __restrict__ gk,
    const float* __restrict__ beta, float* __restrict__ obuf) {
  __shared__ __align__(16) float qs[2][SCAN_T * DK];
  __shared__ __align__(16) float ks[2][SCAN_T * DK];
  __shared__ __align__(16) float vs[2][SCAN_T * SCAN_CW];
  __shared__ float gs[2][SCAN_T];
  __shared__ float bs[2][SCAN_T];

  const int bh = blockIdx.x / SCAN_G;
  const int g  = blockIdx.x % SCAN_G;
  const int b = bh >> 2, h = bh & 3;
  const int tid = threadIdx.x;

  float S[DK];
#pragma unroll
  for (int i = 0; i < DK; ++i) S[i] = 0.f;

  const size_t qcol = (size_t)h * DK;
  const size_t kcol = 768 + (size_t)h * DK;
  const size_t vcol = 1536 + (size_t)h * DV + g * SCAN_CW;
  const size_t rowBase = (size_t)b * NL;
  const float* gkrow = gk + ((size_t)b * NH + h) * NL;
  const float* brow  = beta + ((size_t)b * NH + h) * NL;

  // async-stage one 16-step chunk into buffer `buf`
  auto stage = [&](int buf, int cb) {
    const int l0 = cb * SCAN_T;
    unsigned q0 = lds_off(&qs[buf][0]);
    unsigned k0 = lds_off(&ks[buf][0]);
    unsigned v0 = lds_off(&vs[buf][0]);
    // q,k: 16 rows x 48 float4 = 768 float4 each
#pragma unroll 1
    for (int i = 0; i < 12; ++i) {
      int e  = tid + (i << 6);           // 0..767
      int lt = e / 48;
      int d4 = e - lt * 48;
      size_t rr = (rowBase + l0 + lt) * NCAT;
      async_b128(q0 + e * 16, qkvg + rr + qcol + d4 * 4);
      async_b128(k0 + e * 16, qkvg + rr + kcol + d4 * 4);
    }
    // v: 16 rows x 16 float4 = 256 float4
#pragma unroll 1
    for (int i = 0; i < 4; ++i) {
      int e  = tid + (i << 6);           // 0..255
      int lt = e >> 4;
      int d4 = e & 15;
      async_b128(v0 + e * 16, qkvg + (rowBase + l0 + lt) * NCAT + vcol + d4 * 4);
    }
    if (tid < SCAN_T) {
      gs[buf][tid] = gkrow[l0 + tid];
      bs[buf][tid] = brow[l0 + tid];
    }
  };

  stage(0, 0);
  wait_async0();
  __syncthreads();

  for (int cb = 0; cb < NL / SCAN_T; ++cb) {
    const int cur = cb & 1;
    if (cb + 1 < NL / SCAN_T) stage(cur ^ 1, cb + 1);  // overlap with compute
    const int l0 = cb * SCAN_T;

#pragma unroll 1
    for (int t = 0; t < SCAN_T; ++t) {
      float eg = __expf(gs[cur][t]);
      float bt = bs[cur][t];
      float vt = vs[cur][t * SCAN_CW + tid];
      const float4* k4p = (const float4*)(&ks[cur][t * DK]);
      const float4* q4p = (const float4*)(&qs[cur][t * DK]);
      // pass 1: S *= exp(g); acc = k . S
      float acc = 0.f;
#pragma unroll
      for (int d4 = 0; d4 < 48; ++d4) {
        float4 k4 = k4p[d4];
        S[d4*4+0] *= eg; acc = fmaf(k4.x, S[d4*4+0], acc);
        S[d4*4+1] *= eg; acc = fmaf(k4.y, S[d4*4+1], acc);
        S[d4*4+2] *= eg; acc = fmaf(k4.z, S[d4*4+2], acc);
        S[d4*4+3] *= eg; acc = fmaf(k4.w, S[d4*4+3], acc);
      }
      float u = bt * (vt - acc);
      // pass 2: S += k*u; o = q . S
      float oacc = 0.f;
#pragma unroll
      for (int d4 = 0; d4 < 48; ++d4) {
        float4 k4 = k4p[d4];
        float4 q4 = q4p[d4];
        S[d4*4+0] = fmaf(k4.x, u, S[d4*4+0]); oacc = fmaf(q4.x, S[d4*4+0], oacc);
        S[d4*4+1] = fmaf(k4.y, u, S[d4*4+1]); oacc = fmaf(q4.y, S[d4*4+1], oacc);
        S[d4*4+2] = fmaf(k4.z, u, S[d4*4+2]); oacc = fmaf(q4.z, S[d4*4+2], oacc);
        S[d4*4+3] = fmaf(k4.w, u, S[d4*4+3]); oacc = fmaf(q4.w, S[d4*4+3], oacc);
      }
      obuf[((rowBase + l0 + t) * NH + h) * DV + g * SCAN_CW + tid] = oacc;
    }
    wait_async0();     // next chunk's async copies done (ran behind compute)
    __syncthreads();
  }
}

// ---------------------------------------------------------------------------
// FusedRMSNormSwishGate per (row, head): out = o*rsqrt(mean(o^2)+eps)*w * g*sig(g)
// Emits bf16 for the W_so GEMM.
// ---------------------------------------------------------------------------
__global__ __launch_bounds__(256) void gate_kernel(
    const float* __restrict__ obuf, const float* __restrict__ qkvg,
    const float* __restrict__ gw, __hip_bfloat16* __restrict__ og) {
  int wave = threadIdx.x >> 5, lane = threadIdx.x & 31;
  long long unit = (long long)blockIdx.x * 8 + wave;  // ROWS*4
  int row = (int)(unit >> 2);
  int h = (int)(unit & 3);
  const float* op = obuf + (size_t)row * 1536 + h * DV;
  const float* gp = qkvg + (size_t)row * NCAT + 3072 + h * DV;
  float ov[12];
  float ss = 0.f;
#pragma unroll
  for (int i = 0; i < 12; ++i) { ov[i] = op[lane + i * 32]; ss = fmaf(ov[i], ov[i], ss); }
#pragma unroll
  for (int off = 16; off > 0; off >>= 1) ss += __shfl_xor(ss, off, 32);
  float rms = rsqrtf(ss * (1.f / (float)DV) + 1e-5f);
#pragma unroll
  for (int i = 0; i < 12; ++i) {
    int d = lane + i * 32;
    float gv = gp[d];
    float val = ov[i] * rms * gw[d] * (gv / (1.f + __expf(-gv)));
    og[(size_t)row * 1536 + h * DV + d] = __float2bfloat16(val);
  }
}

// ---------------------------------------------------------------------------
// Launcher
// ---------------------------------------------------------------------------
extern "C" void kernel_launch(void* const* d_in, const int* in_sizes, int n_in,
                              void* d_out, int out_size, void* d_ws, size_t ws_size,
                              hipStream_t stream) {
  (void)in_sizes; (void)n_in; (void)out_size; (void)ws_size;
  const float* x     = (const float*)d_in[0];
  const float* W_in  = (const float*)d_in[1];
  const float* cxw   = (const float*)d_in[2];
  const float* cxb   = (const float*)d_in[3];
  const float* czw   = (const float*)d_in[4];
  const float* czb   = (const float*)d_in[5];
  const float* W_q   = (const float*)d_in[6];
  const float* W_k   = (const float*)d_in[7];
  const float* W_v   = (const float*)d_in[8];
  const float* W_g   = (const float*)d_in[9];
  const float* W_gk  = (const float*)d_in[10];
  const float* W_b   = (const float*)d_in[11];
  const float* b_b   = (const float*)d_in[12];
  const float* A_log = (const float*)d_in[13];
  const float* dtb   = (const float*)d_in[14];
  const float* gnw   = (const float*)d_in[15];
  const float* W_so  = (const float*)d_in[16];
  const float* W_out = (const float*)d_in[17];

  // Workspace partition (~800 MB total)
  char* ws = (char*)d_ws;
  size_t off = 0;
  auto alloc = [&](size_t bytes) -> char* {
    char* p = ws + off;
    off += (bytes + 255) & ~(size_t)255;
    return p;
  };
  __hip_bfloat16* x_bf   = (__hip_bfloat16*)alloc((size_t)ROWS * DM * 2);
  __hip_bfloat16* Win_bf = (__hip_bfloat16*)alloc((size_t)DM * DI * 2);
  __hip_bfloat16* Wcat   = (__hip_bfloat16*)alloc((size_t)DSSM * NCAT * 2);
  __hip_bfloat16* Wso_bf = (__hip_bfloat16*)alloc((size_t)1536 * 1024 * 2);
  __hip_bfloat16* Wout_bf= (__hip_bfloat16*)alloc((size_t)DI * DM * 2);
  float* xz     = (float*)alloc((size_t)ROWS * DI * 4);
  float* xs     = (float*)alloc((size_t)ROWS * DSSM * 4);
  __hip_bfloat16* xs_bf  = (__hip_bfloat16*)alloc((size_t)ROWS * DSSM * 2);
  float* qkvg   = (float*)alloc((size_t)ROWS * NCAT * 4);
  float* gkbuf  = (float*)alloc((size_t)NB * NH * NL * 4);
  float* btbuf  = (float*)alloc((size_t)NB * NH * NL * 4);
  float* obuf   = (float*)alloc((size_t)ROWS * 1536 * 4);
  __hip_bfloat16* og_bf  = (__hip_bfloat16*)alloc((size_t)ROWS * 1536 * 2);
  __hip_bfloat16* ycat   = (__hip_bfloat16*)alloc((size_t)ROWS * DI * 2);

  // 1) bf16 casts of activations and weights
  cast_f32_to_bf16_slice<<<(ROWS * DM + 255) / 256, 256, 0, stream>>>(x, x_bf, (long long)ROWS * DM, DM, DM, 0);
  cast_f32_to_bf16_slice<<<(DM * DI + 255) / 256, 256, 0, stream>>>(W_in, Win_bf, (long long)DM * DI, DI, DI, 0);
  cast_f32_to_bf16_slice<<<(DSSM * 768 + 255) / 256, 256, 0, stream>>>(W_q, Wcat, (long long)DSSM * 768, 768, NCAT, 0);
  cast_f32_to_bf16_slice<<<(DSSM * 768 + 255) / 256, 256, 0, stream>>>(W_k, Wcat, (long long)DSSM * 768, 768, NCAT, 768);
  cast_f32_to_bf16_slice<<<(DSSM * 1536 + 255) / 256, 256, 0, stream>>>(W_v, Wcat, (long long)DSSM * 1536, 1536, NCAT, 1536);
  cast_f32_to_bf16_slice<<<(DSSM * 1536 + 255) / 256, 256, 0, stream>>>(W_g, Wcat, (long long)DSSM * 1536, 1536, NCAT, 3072);
  cast_f32_to_bf16_slice<<<(1536 * 1024 + 255) / 256, 256, 0, stream>>>(W_so, Wso_bf, (long long)1536 * 1024, 1024, 1024, 0);
  cast_f32_to_bf16_slice<<<(DI * DM + 255) / 256, 256, 0, stream>>>(W_out, Wout_bf, (long long)DI * DM, DM, DM, 0);

  // 2) xz = x @ W_in   [16384 x 2048], K=1024
  gemm_bf16_wmma<false><<<dim3(ROWS / 128, DI / 128), 256, 0, stream>>>(
      (const u16*)x_bf, (const u16*)Win_bf, ROWS, DI, DM, xz, DI, 0);

  // 3) depthwise conv + SiLU: x branch -> xs (f32 + bf16); z branch -> bf16 into ycat[:,1024:]
  conv1d_silu<<<(ROWS * 1024 + 255) / 256, 256, 0, stream>>>(xz, 0, cxw, cxb, xs, xs_bf, DSSM, 0);
  conv1d_silu<<<(ROWS * 1024 + 255) / 256, 256, 0, stream>>>(xz, 1024, czw, czb, nullptr, ycat, DI, 1024);

  // 4) qkvg = xs @ [Wq|Wk|Wv|Wg]   [16384 x 4608], K=1024
  gemm_bf16_wmma<false><<<dim3(ROWS / 128, NCAT / 128), 256, 0, stream>>>(
      (const u16*)xs_bf, (const u16*)Wcat, ROWS, NCAT, DSSM, qkvg, NCAT, 0);

  // 5) gk / beta (f32 dots)
  gk_beta_kernel<<<ROWS, 256, 0, stream>>>(xs, W_gk, W_b, dtb, b_b, A_log, gkbuf, btbuf);

  // 6) L2-normalize q,k in place
  l2norm_qk<<<ROWS, 256, 0, stream>>>(qkvg);

  // 7) gated delta rule scan -> obuf [ROWS, H*DV]
  gated_delta_scan<<<NB * NH * SCAN_G, 64, 0, stream>>>(qkvg, gkbuf, btbuf, obuf);

  // 8) RMSNorm + swish gate -> bf16
  gate_kernel<<<ROWS / 2, 256, 0, stream>>>(obuf, qkvg, gnw, og_bf);

  // 9) y = og @ W_so -> bf16 into ycat[:,0:1024]
  gemm_bf16_wmma<true><<<dim3(ROWS / 128, 1024 / 128), 256, 0, stream>>>(
      (const u16*)og_bf, (const u16*)Wso_bf, ROWS, 1024, 1536, ycat, DI, 0);

  // 10) out = [y, zs] @ W_out -> f32 d_out
  gemm_bf16_wmma<false><<<dim3(ROWS / 128, DM / 128), 256, 0, stream>>>(
      (const u16*)ycat, (const u16*)Wout_bf, ROWS, DM, DI, d_out, DM, 0);
}
